// IDRM_ANN_81389630259799
// MI455X (gfx1250) — compile-verified, matrix operand order checked by reading, not run on hardware
//
#include <hip/hip_runtime.h>

// ---------------------------------------------------------------------------
// Problem dimensions (fixed by the reference)
// ---------------------------------------------------------------------------
#define BATCH 16384
#define NIN   256
#define NHID  1024
#define NOUT  256

typedef __attribute__((ext_vector_type(16))) __bf16 bf16x16;
typedef __attribute__((ext_vector_type(8)))  float  f32x8;

// ---------------------------------------------------------------------------
// bf16 helpers (manual RNE conversion; no arithmetic on __bf16 itself)
// ---------------------------------------------------------------------------
__device__ __forceinline__ unsigned short f2bf(float f) {
    unsigned int u = __float_as_uint(f);
    unsigned int r = (u + 0x7FFFu + ((u >> 16) & 1u)) >> 16;
    return (unsigned short)r;
}
__device__ __forceinline__ float bf2f(unsigned short h) {
    return __uint_as_float(((unsigned int)h) << 16);
}

// Native gfx1250 transcendental tanh (TRANS32 class) with libm fallback.
__device__ __forceinline__ float fast_tanh(float v) {
#if __has_builtin(__builtin_amdgcn_tanh_f32)
    return __builtin_amdgcn_tanh_f32(v);
#elif __has_builtin(__builtin_amdgcn_tanhf)
    return __builtin_amdgcn_tanhf(v);
#else
    return tanhf(v);
#endif
}

// ---------------------------------------------------------------------------
// WMMA fragment load for 16-bit A(16x32, MxK) / B(32x16, KxN from row-major NxK).
// Lane l holds row (l&15); K chunks [kb, kb+8) and [kb+16, kb+24), kb = 8*(l>=16).
// Two global_load_b128 per lane (one address, offset:32 pair).
// ---------------------------------------------------------------------------
__device__ __forceinline__ bf16x16 load_frag(const unsigned short* base, int ld,
                                             int row, int k, int lane) {
    const int r  = row + (lane & 15);
    const int kb = k + ((lane >> 4) << 3);
    const unsigned short* p = base + (size_t)r * (size_t)ld + kb;
    union { uint4 u[2]; bf16x16 v; } cv;
    cv.u[0] = *reinterpret_cast<const uint4*>(p);       // K = kb .. kb+7
    cv.u[1] = *reinterpret_cast<const uint4*>(p + 16);  // K = kb+16 .. kb+23
    return cv.v;
}

__device__ __forceinline__ f32x8 wmma_bf16(bf16x16 a, bf16x16 b, f32x8 c) {
    // 8 args: (neg_a, A, neg_b, B, c_mod, C, reuse_a, reuse_b)
    return __builtin_amdgcn_wmma_f32_16x16x32_bf16(false, a, false, b,
                                                   (short)0, c, false, false);
}

// One K-step (K += 32) of the 64x64 wave tile: 8 fragment loads, 16 WMMAs.
__device__ __forceinline__ void mma_step(const unsigned short* __restrict__ A,
                                         const unsigned short* __restrict__ B,
                                         int ld, int m0, int n0, int k, int lane,
                                         f32x8 acc[4][4]) {
    bf16x16 a[4], b[4];
    #pragma unroll
    for (int i = 0; i < 4; ++i) a[i] = load_frag(A, ld, m0 + i * 16, k, lane);
    #pragma unroll
    for (int j = 0; j < 4; ++j) b[j] = load_frag(B, ld, n0 + j * 16, k, lane);
    #pragma unroll
    for (int i = 0; i < 4; ++i)
        #pragma unroll
        for (int j = 0; j < 4; ++j)
            acc[i][j] = wmma_bf16(a[i], b[j], acc[i][j]);
}

// ---------------------------------------------------------------------------
// Elementwise prep kernels
// ---------------------------------------------------------------------------
__global__ void prep_x_kernel(const float* __restrict__ x,
                              unsigned short* __restrict__ out, int n) {
    int i = blockIdx.x * blockDim.x + threadIdx.x;
    if (i < n) out[i] = f2bf(fabsf(x[i]) + 0.1f);
}

__global__ void cvt_bf16_kernel(const float* __restrict__ w,
                                unsigned short* __restrict__ out, int n) {
    int i = blockIdx.x * blockDim.x + threadIdx.x;
    if (i < n) out[i] = f2bf(w[i]);
}

// ---------------------------------------------------------------------------
// Dual-K WMMA GEMM:  acc = A1(bf16, MxK1) @ B1(bf16, N x K1)^T
//                        [+ A2(bf16, MxK2) @ B2(bf16, N x K2)^T]
// Block: 256 threads = 8 waves (4 along M, 2 along N).
// Block tile 256x128; wave tile 64x64 (4x4 WMMA tiles, 128 acc VGPRs).
// K loop unrolled by 64 (two 32-steps) for load/WMMA overlap.
// Epilogue MODE:
//   0: outF = acc (+bias)                       (final y, f32)
//   1: outB = bf16(tanh(acc+bias))              (s0, g1, z1)
//   2: outB = bf16(bf2f(e0) * tanh(acc+bias))   (sr = s0 * r1, fused)
//   3: h=tanh(acc+bias); outB = bf16((1-g)*h + z*s0), g=e0,z=e1,s0=e2 (s1, fused)
// ---------------------------------------------------------------------------
template <int HAS2, int BIAS, int MODE>
__global__ __launch_bounds__(256)
void gemm_wmma(const unsigned short* __restrict__ A1,
               const unsigned short* __restrict__ B1, int K1,
               const unsigned short* __restrict__ A2,
               const unsigned short* __restrict__ B2, int K2,
               const float* __restrict__ bias,
               const unsigned short* __restrict__ e0,
               const unsigned short* __restrict__ e1,
               const unsigned short* __restrict__ e2,
               float* __restrict__ outF,
               unsigned short* __restrict__ outB, int N) {
    const int lane = threadIdx.x & 31;
    const int wave = threadIdx.x >> 5;   // 0..7
    const int wm   = wave & 3;           // 4 waves along M
    const int wn   = wave >> 2;          // 2 waves along N
    const int m0   = blockIdx.x * 256 + wm * 64;   // wave covers 64 rows
    const int n0   = blockIdx.y * 128 + wn * 64;   // wave covers 64 cols

    f32x8 acc[4][4];
    #pragma unroll
    for (int i = 0; i < 4; ++i)
        #pragma unroll
        for (int j = 0; j < 4; ++j)
            acc[i][j] = f32x8{};

    for (int k = 0; k < K1; k += 64) {
        mma_step(A1, B1, K1, m0, n0, k,      lane, acc);
        mma_step(A1, B1, K1, m0, n0, k + 32, lane, acc);
    }
    if (HAS2) {
        for (int k = 0; k < K2; k += 64) {
            mma_step(A2, B2, K2, m0, n0, k,      lane, acc);
            mma_step(A2, B2, K2, m0, n0, k + 32, lane, acc);
        }
    }

    // Epilogue. C/D layout: VGPR r -> row r + 8*(lane>=16), col = lane&15.
    const int dm8 = (lane >> 4) << 3;
    const int dn  = lane & 15;
    #pragma unroll
    for (int i = 0; i < 4; ++i) {
        #pragma unroll
        for (int j = 0; j < 4; ++j) {
            const int col = n0 + j * 16 + dn;
            const float bv = BIAS ? bias[col] : 0.0f;
            #pragma unroll
            for (int r = 0; r < 8; ++r) {
                const int row = m0 + i * 16 + dm8 + r;
                const size_t idx = (size_t)row * (size_t)N + col;
                float v = acc[i][j][r] + bv;
                if (MODE == 0) {
                    outF[idx] = v;
                } else if (MODE == 1) {
                    outB[idx] = f2bf(fast_tanh(v));
                } else if (MODE == 2) {
                    outB[idx] = f2bf(bf2f(e0[idx]) * fast_tanh(v));
                } else {
                    float h = fast_tanh(v);
                    float g = bf2f(e0[idx]);
                    float z = bf2f(e1[idx]);
                    float s = bf2f(e2[idx]);
                    outB[idx] = f2bf((1.0f - g) * h + z * s);
                }
            }
        }
    }
}

// ---------------------------------------------------------------------------
// Host-side orchestration
// ---------------------------------------------------------------------------
static inline char* ws_alloc(char*& ws, size_t bytes) {
    char* p = ws;
    ws += (bytes + 255) & ~(size_t)255;
    return p;
}

extern "C" void kernel_launch(void* const* d_in, const int* in_sizes, int n_in,
                              void* d_out, int out_size, void* d_ws, size_t ws_size,
                              hipStream_t stream) {
    (void)in_sizes; (void)n_in; (void)out_size; (void)ws_size;

    const float* x     = (const float*)d_in[0];
    const float* w0_w  = (const float*)d_in[1];
    const float* w0_b  = (const float*)d_in[2];
    const float* ug1   = (const float*)d_in[3];
    const float* wg1_w = (const float*)d_in[4];
    const float* wg1_b = (const float*)d_in[5];
    const float* uz1   = (const float*)d_in[6];
    const float* wz1_w = (const float*)d_in[7];
    const float* wz1_b = (const float*)d_in[8];
    const float* ur1   = (const float*)d_in[9];
    const float* wr1_w = (const float*)d_in[10];
    const float* wr1_b = (const float*)d_in[11];
    const float* uh1   = (const float*)d_in[12];
    const float* wh1_w = (const float*)d_in[13];
    const float* wh1_b = (const float*)d_in[14];
    const float* w4    = (const float*)d_in[27];
    float* y = (float*)d_out;

    // Workspace carve-out (~190 MB)
    char* ws = (char*)d_ws;
    unsigned short* xa  = (unsigned short*)ws_alloc(ws, (size_t)BATCH * NIN * 2);
    unsigned short* w0b = (unsigned short*)ws_alloc(ws, (size_t)NHID * NIN * 2);
    unsigned short* ugb = (unsigned short*)ws_alloc(ws, (size_t)NHID * NIN * 2);
    unsigned short* uzb = (unsigned short*)ws_alloc(ws, (size_t)NHID * NIN * 2);
    unsigned short* urb = (unsigned short*)ws_alloc(ws, (size_t)NHID * NIN * 2);
    unsigned short* uhb = (unsigned short*)ws_alloc(ws, (size_t)NHID * NIN * 2);
    unsigned short* wgb = (unsigned short*)ws_alloc(ws, (size_t)NHID * NHID * 2);
    unsigned short* wzb = (unsigned short*)ws_alloc(ws, (size_t)NHID * NHID * 2);
    unsigned short* wrb = (unsigned short*)ws_alloc(ws, (size_t)NHID * NHID * 2);
    unsigned short* whb = (unsigned short*)ws_alloc(ws, (size_t)NHID * NHID * 2);
    unsigned short* w4b = (unsigned short*)ws_alloc(ws, (size_t)NOUT * NHID * 2);
    unsigned short* s0b = (unsigned short*)ws_alloc(ws, (size_t)BATCH * NHID * 2);
    unsigned short* g1b = (unsigned short*)ws_alloc(ws, (size_t)BATCH * NHID * 2);
    unsigned short* z1b = (unsigned short*)ws_alloc(ws, (size_t)BATCH * NHID * 2);
    unsigned short* srb = (unsigned short*)ws_alloc(ws, (size_t)BATCH * NHID * 2);
    unsigned short* s1b = (unsigned short*)ws_alloc(ws, (size_t)BATCH * NHID * 2);

    // --- 1) xa = bf16(|x| + 0.1) ---------------------------------------------
    {
        int n = BATCH * NIN;
        prep_x_kernel<<<(n + 255) / 256, 256, 0, stream>>>(x, xa, n);
    }
    // --- 2) weights -> bf16 ---------------------------------------------------
    {
        struct { const float* src; unsigned short* dst; int n; } cv[] = {
            { w0_w,  w0b, NHID * NIN  }, { ug1, ugb, NHID * NIN },
            { uz1,   uzb, NHID * NIN  }, { ur1, urb, NHID * NIN },
            { uh1,   uhb, NHID * NIN  }, { wg1_w, wgb, NHID * NHID },
            { wz1_w, wzb, NHID * NHID }, { wr1_w, wrb, NHID * NHID },
            { wh1_w, whb, NHID * NHID }, { w4,  w4b, NOUT * NHID },
        };
        for (auto& c : cv)
            cvt_bf16_kernel<<<(c.n + 255) / 256, 256, 0, stream>>>(c.src, c.dst, c.n);
    }

    const dim3 blk(256);
    const dim3 grid_h(BATCH / 256, NHID / 128);  // N = 1024 -> (64, 8)
    const dim3 grid_o(BATCH / 256, NOUT / 128);  // N = 256  -> (64, 2)

    // --- 3) s0 = tanh(xa @ w0^T + b0) -> bf16 --------------------------------
    gemm_wmma<0, 1, 1><<<grid_h, blk, 0, stream>>>(
        xa, w0b, NIN, nullptr, nullptr, 0, w0_b,
        nullptr, nullptr, nullptr, nullptr, s0b, NHID);

    // --- 4) g1 = tanh(xa @ ug^T + s0 @ wg^T + bg) -> bf16 --------------------
    gemm_wmma<1, 1, 1><<<grid_h, blk, 0, stream>>>(
        xa, ugb, NIN, s0b, wgb, NHID, wg1_b,
        nullptr, nullptr, nullptr, nullptr, g1b, NHID);

    // --- 5) z1 = tanh(xa @ uz^T + s0 @ wz^T + bz) -> bf16 --------------------
    gemm_wmma<1, 1, 1><<<grid_h, blk, 0, stream>>>(
        xa, uzb, NIN, s0b, wzb, NHID, wz1_b,
        nullptr, nullptr, nullptr, nullptr, z1b, NHID);

    // --- 6) sr = s0 * tanh(xa @ ur^T + s0 @ wr^T + br) -> bf16 (r1 fused) ----
    gemm_wmma<1, 1, 2><<<grid_h, blk, 0, stream>>>(
        xa, urb, NIN, s0b, wrb, NHID, wr1_b,
        s0b, nullptr, nullptr, nullptr, srb, NHID);

    // --- 7) h1 = tanh(xa @ uh^T + sr @ wh^T + bh); s1 = (1-g1)h1 + z1*s0 -----
    gemm_wmma<1, 1, 3><<<grid_h, blk, 0, stream>>>(
        xa, uhb, NIN, srb, whb, NHID, wh1_b,
        g1b, z1b, s0b, nullptr, s1b, NHID);

    // --- 8) y = s1 @ w4^T -> f32 out -----------------------------------------
    gemm_wmma<0, 0, 0><<<grid_o, blk, 0, stream>>>(
        s1b, w4b, NHID, nullptr, nullptr, 0, nullptr,
        nullptr, nullptr, nullptr, y, nullptr, NOUT);
}
